// PulseModel_64424509440476
// MI455X (gfx1250) — matrix operand order repaired and checked by reading, not compiled, not run on hardware
//
#include <hip/hip_runtime.h>

typedef __attribute__((ext_vector_type(16))) __bf16 v16bf;
typedef __attribute__((ext_vector_type(8)))  __bf16 v8bf;
typedef __attribute__((ext_vector_type(8)))  float  v8f;

#define Bc   2
#define Sc   1024
#define Dc   1024
#define Hc   16
#define Nc   32
#define DSc  1024
#define Fic  4096
#define Lc   8
#define BSc  (Bc*Sc)      // 2048
#define BNc  (Bc*Nc)      // 64
#define D3c  (3*Dc)       // 3072
#define D2c  (2*DSc)      // 2048
#define DHc  64

// ---------------- helpers ----------------

// CDNA5 async copy: global (fp32, 16B) -> LDS, tracked by ASYNCcnt.
// Generic pointers to __shared__ carry the LDS byte offset in their low 32 bits.
__device__ __forceinline__ void async_g2l_b128(const float* g, void* lds){
  unsigned loff = (unsigned)(unsigned long long)(uintptr_t)lds;
  asm volatile("global_load_async_to_lds_b128 %0, %1, off"
               :: "v"(loff), "v"(g) : "memory");
}
__device__ __forceinline__ void wait_async0(){
  asm volatile("s_wait_asynccnt 0" ::: "memory");
}

// A-fragment (16x32 bf16): element i -> k = 2*off + i (16 contiguous)
__device__ __forceinline__ v16bf load_a_f32(const float* p, int off){
  v16bf r;
#pragma unroll
  for (int i = 0; i < 8; ++i){
    r[i]     = (__bf16)p[2 * off + i];
    r[i + 8] = (__bf16)p[2 * off + 8 + i];
  }
  return r;
}

// B-fragment (32x16 bf16): element i -> k = kbase + i (16 contiguous)
__device__ __forceinline__ v16bf load_b16_f32(const float* p){
  v16bf r;
#pragma unroll
  for (int i = 0; i < 16; ++i) r[i] = (__bf16)p[i];
  return r;
}

__device__ __forceinline__ v16bf load_b16_strided(const float* p, int stride){
  v16bf r;
#pragma unroll
  for (int i = 0; i < 16; ++i) r[i] = (__bf16)p[(size_t)i * stride];
  return r;
}

__device__ __forceinline__ v16bf mk16(v8bf lo, v8bf hi){
  v16bf r;
#pragma unroll
  for (int i = 0; i < 8; ++i){ r[i] = lo[i]; r[i + 8] = hi[i]; }
  return r;
}

__device__ __forceinline__ float act_apply(float v, int act){
  if (act == 1) return 0.5f * v * (1.0f + erff(v * 0.70710678118654752f)); // exact GELU
  if (act == 2) return 1.0f / (1.0f + __expf(-v));                        // sigmoid
  if (act == 3) return tanhf(v);                                          // tanh
  return v;
}

// ---------------- generic WMMA GEMM: C[M,N] = act(A[M,K] @ B[K,N] + bias) + resid ----------------
// fp32 tiles staged global->LDS with GLOBAL_LOAD_ASYNC_TO_LDS_B128 (ASYNCcnt),
// converted once to bf16 in LDS, then consumed as WMMA fragments (ds_load_b128).

__global__ __launch_bounds__(256) void k_gemm(
    const float* __restrict__ A, int lda,
    const float* __restrict__ Bm, int ldb,
    const float* __restrict__ bias,
    const float* __restrict__ resid,
    float* __restrict__ C, int ldc,
    int K, int act)
{
  __shared__ __align__(16) float  fA[64][36];   // raw fp32 A tile [m][k] (row 144B)
  __shared__ __align__(16) float  fB[32][68];   // raw fp32 B tile [k][n] (row 272B)
  __shared__ __align__(16) __bf16 sA[64][40];   // bf16 [m][k]
  __shared__ __align__(16) __bf16 sB[64][40];   // bf16 [n][k] (transposed)
  const int tid  = threadIdx.x;
  const int lane = tid & 31;
  const int wid  = tid >> 5;
  const int m0 = blockIdx.y * 64;
  const int n0 = blockIdx.x * 64;
  const int wm = (wid & 3) * 16;       // wave row offset in tile
  const int wn = (wid >> 2) * 32;      // wave col offset in tile

  const int sm = tid >> 2;             // A: row 0..63
  const int sk = (tid & 3) * 8;        // A: k seg {0,8,16,24}
  const int bk = tid >> 3;             // B: k row 0..31
  const int bn = (tid & 7) * 8;        // B: n seg {0..56}

  v8f acc0 = {0.f,0.f,0.f,0.f,0.f,0.f,0.f,0.f};
  v8f acc1 = acc0;

  const float* apg = A  + (size_t)(m0 + sm) * lda + sk;
  const float* bpg = Bm + (size_t)bk * ldb + n0 + bn;

  for (int k0 = 0; k0 < K; k0 += 32){
    // ---- async stage raw fp32 tiles into LDS ----
    async_g2l_b128(apg + k0,     &fA[sm][sk]);
    async_g2l_b128(apg + k0 + 4, &fA[sm][sk + 4]);
    const float* bpp = bpg + (size_t)k0 * ldb;
    async_g2l_b128(bpp,     &fB[bk][bn]);
    async_g2l_b128(bpp + 4, &fB[bk][bn + 4]);
    wait_async0();
    __syncthreads();
    // ---- convert fp32 -> bf16 (B transposed to [n][k]) ----
#pragma unroll
    for (int j = 0; j < 8; ++j) sA[sm][sk + j] = (__bf16)fA[sm][sk + j];
#pragma unroll
    for (int j = 0; j < 8; ++j) sB[bn + j][bk] = (__bf16)fB[bk][bn + j];
    __syncthreads();
    // ---- WMMA fragments ----
    const int row = wm + (lane & 15);
    const int off = (lane < 16) ? 0 : 8;
    v16bf af = mk16(*(const v8bf*)&sA[row][off],
                    *(const v8bf*)&sA[row][16 + off]);
    const int kb  = (lane < 16) ? 0 : 16;
    const int nc0 = wn + (lane & 15);
    v16bf bf0 = mk16(*(const v8bf*)&sB[nc0][off],
                     *(const v8bf*)&sB[nc0][16 + off]);
    v16bf bf1 = mk16(*(const v8bf*)&sB[nc0 + 16][off],
                     *(const v8bf*)&sB[nc0 + 16][16 + off]);
    acc0 = __builtin_amdgcn_wmma_f32_16x16x32_bf16(false, af, false, bf0, (short)0, acc0, false, false);
    acc1 = __builtin_amdgcn_wmma_f32_16x16x32_bf16(false, af, false, bf1, (short)0, acc1, false, false);
    __syncthreads();
  }

#pragma unroll
  for (int s = 0; s < 2; ++s){
    v8f acc = s ? acc1 : acc0;
    const int nc = n0 + wn + s * 16 + (lane & 15);
    const float bv = bias ? bias[nc] : 0.0f;
#pragma unroll
    for (int r = 0; r < 8; ++r){
      const int mr = m0 + wm + ((lane < 16) ? r : r + 8);
      float v = act_apply(acc[r] + bv, act);
      if (resid) v += resid[(size_t)mr * ldc + nc];
      C[(size_t)mr * ldc + nc] = v;
    }
  }
}

// ---------------- LayerNorm (row length Dc=1024, 256 thr, 4 elem/thr) ----------------

__device__ __forceinline__ void ln_reduce_write(float e0, float e1, float e2, float e3,
    const float* __restrict__ w, const float* __restrict__ b,
    float* __restrict__ out, int row, int d)
{
  __shared__ float s1[256], s2[256];
  int tid = threadIdx.x;
  s1[tid] = e0 + e1 + e2 + e3;
  s2[tid] = e0*e0 + e1*e1 + e2*e2 + e3*e3;
  __syncthreads();
  for (int o = 128; o > 0; o >>= 1){
    if (tid < o){ s1[tid] += s1[tid + o]; s2[tid] += s2[tid + o]; }
    __syncthreads();
  }
  float mean = s1[0] * (1.0f / Dc);
  float var  = s2[0] * (1.0f / Dc) - mean * mean;
  float inv  = rsqrtf(var + 1e-6f);
  float* op = out + (size_t)row * Dc + d;
  op[0] = (e0 - mean) * inv * w[d+0] + b[d+0];
  op[1] = (e1 - mean) * inv * w[d+1] + b[d+1];
  op[2] = (e2 - mean) * inv * w[d+2] + b[d+2];
  op[3] = (e3 - mean) * inv * w[d+3] + b[d+3];
}

__global__ __launch_bounds__(256) void k_embed_ln(const int* __restrict__ ids,
    const float* __restrict__ we, const float* __restrict__ pe,
    const float* __restrict__ w, const float* __restrict__ b,
    float* __restrict__ out)
{
  int row = blockIdx.x;
  int s = row % Sc;
  int tok = ids[row];
  int d = threadIdx.x * 4;
  float4 a = *(const float4*)(we + (size_t)tok * Dc + d);
  float4 c = *(const float4*)(pe + (size_t)s   * Dc + d);
  ln_reduce_write(a.x + c.x, a.y + c.y, a.z + c.z, a.w + c.w, w, b, out, row, d);
}

__global__ __launch_bounds__(256) void k_ln(const float* __restrict__ in,
    const float* __restrict__ w, const float* __restrict__ b,
    float* __restrict__ out)
{
  int row = blockIdx.x;
  int d = threadIdx.x * 4;
  float4 a = *(const float4*)(in + (size_t)row * Dc + d);
  ln_reduce_write(a.x, a.y, a.z, a.w, w, b, out, row, d);
}

// ---------------- flash causal MHA: one wave per (b, h, 16-query tile) ----------------

__global__ __launch_bounds__(256) void k_attn(const float* __restrict__ qkv,
                                              float* __restrict__ out)
{
  __shared__ __align__(16) __bf16 sP[8][16][40];  // per-wave P transpose slab
  const int tid  = threadIdx.x;
  const int lane = tid & 31;
  const int wid  = tid >> 5;
  const int g  = blockIdx.x * 8 + wid;     // 0..2047
  const int qt = g & 63;                   // S/16 = 64
  const int hh = (g >> 6) & 15;
  const int bb = g >> 10;
  const int qbase = qt * 16;

  const int ln15 = lane & 15;
  const int off  = (lane < 16) ? 0 : 8;
  const int kb   = (lane < 16) ? 0 : 16;

  const float* qp = qkv + ((size_t)(bb * Sc + qbase + ln15)) * D3c + hh * DHc;
  v16bf qf0 = load_a_f32(qp, off);          // dh 0..31
  v16bf qf1 = load_a_f32(qp + 32, off);     // dh 32..63

  float mrow[8], lrow[8];
  v8f zero = {0.f,0.f,0.f,0.f,0.f,0.f,0.f,0.f};
  v8f oacc[4];
#pragma unroll
  for (int r = 0; r < 8; ++r){ mrow[r] = -1e30f; lrow[r] = 0.0f; }
#pragma unroll
  for (int d = 0; d < 4; ++d) oacc[d] = zero;

  for (int kt = 0; kt < qbase + 16; kt += 32){
    v8f sc0 = zero, sc1 = zero;
    {
      const float* kp0 = qkv + ((size_t)(bb * Sc + kt + ln15)) * D3c + Dc + hh * DHc + kb;
      sc0 = __builtin_amdgcn_wmma_f32_16x16x32_bf16(false, qf0, false, load_b16_f32(kp0),      (short)0, sc0, false, false);
      sc0 = __builtin_amdgcn_wmma_f32_16x16x32_bf16(false, qf1, false, load_b16_f32(kp0 + 32), (short)0, sc0, false, false);
      const float* kp1 = kp0 + (size_t)16 * D3c;
      sc1 = __builtin_amdgcn_wmma_f32_16x16x32_bf16(false, qf0, false, load_b16_f32(kp1),      (short)0, sc1, false, false);
      sc1 = __builtin_amdgcn_wmma_f32_16x16x32_bf16(false, qf1, false, load_b16_f32(kp1 + 32), (short)0, sc1, false, false);
    }
    __builtin_amdgcn_wave_barrier();
#pragma unroll
    for (int r = 0; r < 8; ++r){
      const int srow = qbase + ((lane < 16) ? r : r + 8);
      const int key0 = kt + ln15;
      float v0 = sc0[r] * 0.125f; if (key0 > srow)      v0 = -1e30f;
      float v1 = sc1[r] * 0.125f; if (key0 + 16 > srow) v1 = -1e30f;
      float mx = fmaxf(v0, v1);
#pragma unroll
      for (int msk = 1; msk < 16; msk <<= 1) mx = fmaxf(mx, __shfl_xor(mx, msk, 32));
      float mn   = fmaxf(mrow[r], mx);
      float corr = __expf(mrow[r] - mn);
      float p0 = __expf(v0 - mn);
      float p1 = __expf(v1 - mn);
      float ts = p0 + p1;
#pragma unroll
      for (int msk = 1; msk < 16; msk <<= 1) ts += __shfl_xor(ts, msk, 32);
      lrow[r] = lrow[r] * corr + ts;
      mrow[r] = mn;
#pragma unroll
      for (int d = 0; d < 4; ++d) oacc[d][r] *= corr;
      const int mloc = (lane < 16) ? r : r + 8;
      sP[wid][mloc][ln15]      = (__bf16)p0;
      sP[wid][mloc][16 + ln15] = (__bf16)p1;
    }
    __builtin_amdgcn_wave_barrier();
    v16bf pf = mk16(*(const v8bf*)&sP[wid][ln15][2 * off],
                    *(const v8bf*)&sP[wid][ln15][2 * off + 8]);
#pragma unroll
    for (int d = 0; d < 4; ++d){
      const float* vp = qkv + ((size_t)(bb * Sc + kt + kb)) * D3c + 2 * Dc + hh * DHc + d * 16 + ln15;
      oacc[d] = __builtin_amdgcn_wmma_f32_16x16x32_bf16(false, pf, false, load_b16_strided(vp, D3c), (short)0, oacc[d], false, false);
    }
    __builtin_amdgcn_wave_barrier();
  }

#pragma unroll
  for (int d = 0; d < 4; ++d){
    const int nc = hh * DHc + d * 16 + ln15;
#pragma unroll
    for (int r = 0; r < 8; ++r){
      const int mloc = (lane < 16) ? r : r + 8;
      out[((size_t)(bb * Sc + qbase + mloc)) * Dc + nc] = oacc[d][r] / lrow[r];
    }
  }
}

// ---------------- sparse top-8 state attention: one wave per (b, h, s) ----------------

__global__ __launch_bounds__(256) void k_state_attn(
    const float* __restrict__ q2, const float* __restrict__ ks,
    const float* __restrict__ vs, float* __restrict__ so)
{
  const int tid  = threadIdx.x;
  const int lane = tid & 31;
  const int wid  = tid >> 5;
  const int g  = blockIdx.x * 8 + wid;     // g = (b*H + h)*S + s
  const int s  = g & (Sc - 1);
  const int hh = (g >> 10) & (Hc - 1);
  const int bb = g >> 14;

  const float* qp = q2 + ((size_t)(bb * Sc + s)) * Dc + hh * DHc;
  const float* kp = ks + ((size_t)(bb * Nc + lane)) * Dc + hh * DHc;
  float sc = 0.0f;
#pragma unroll 8
  for (int d = 0; d < DHc; ++d) sc += qp[d] * kp[d];
  sc *= 0.125f;

  float bvv[8]; int bii[8];
  float v = sc;
#pragma unroll
  for (int k = 0; k < 8; ++k){
    float mv = v; int mi = lane;
#pragma unroll
    for (int o = 16; o > 0; o >>= 1){
      float ov = __shfl_xor(mv, o, 32);
      int   oi = __shfl_xor(mi, o, 32);
      if (ov > mv || (ov == mv && oi < mi)){ mv = ov; mi = oi; }
    }
    bvv[k] = mv; bii[k] = mi;
    if (lane == mi) v = -3.0e38f;
  }
  float mmax = bvv[0];
  float sum = 0.0f;
#pragma unroll
  for (int k = 0; k < 8; ++k) sum += __expf(bvv[k] - mmax);
  float a0 = 0.0f, a1 = 0.0f;
#pragma unroll
  for (int k = 0; k < 8; ++k){
    float w = __expf(bvv[k] - mmax) / sum;
    const float* vp = vs + ((size_t)(bb * Nc + bii[k])) * Dc + hh * DHc;
    a0 += w * vp[lane];
    a1 += w * vp[lane + 32];
  }
  float* op = so + ((size_t)(bb * Sc + s)) * Dc + hh * DHc;
  op[lane]      = a0;
  op[lane + 32] = a1;
}

// ---------------- small elementwise kernels ----------------

__global__ __launch_bounds__(256) void k_bcast_st(float* __restrict__ st, const float* __restrict__ init){
  int i = blockIdx.x * 256 + threadIdx.x;
  st[i] = init[i & (Nc * DSc - 1)];
}

__global__ __launch_bounds__(256) void k_mean_seq(const float* __restrict__ h, float* __restrict__ si){
  int i = blockIdx.x * 256 + threadIdx.x;      // over B*D
  int b = i >> 10; int d = i & (Dc - 1);
  const float* p = h + ((size_t)b * Sc) * Dc + d;
  float s = 0.0f;
  for (int t = 0; t < Sc; ++t) s += p[(size_t)t * Dc];
  si[i] = s * (1.0f / Sc);
}

__global__ __launch_bounds__(256) void k_cat(const float* __restrict__ st, const float* __restrict__ si, float* __restrict__ cat){
  int i = blockIdx.x * 256 + threadIdx.x;      // over BNc*D2c
  int row = i / D2c; int j = i - row * D2c;
  int b = row / Nc;
  cat[i] = (j < DSc) ? st[(size_t)row * DSc + j] : si[(size_t)b * Dc + (j - DSc)];
}

__global__ __launch_bounds__(256) void k_cat2(const float* __restrict__ st, const float* __restrict__ si, const float* __restrict__ rr, float* __restrict__ cat){
  int i = blockIdx.x * 256 + threadIdx.x;
  int row = i / D2c; int j = i - row * D2c;
  int b = row / Nc;
  cat[i] = (j < DSc) ? rr[(size_t)row * DSc + j] * st[(size_t)row * DSc + j]
                     : si[(size_t)b * Dc + (j - DSc)];
}

__global__ __launch_bounds__(256) void k_gru(float* __restrict__ st, const float* __restrict__ z, const float* __restrict__ hc){
  int i = blockIdx.x * 256 + threadIdx.x;
  float zz = z[i];
  st[i] = (1.0f - zz) * st[i] + zz * hc[i];
}

__global__ __launch_bounds__(256) void k_copy(float* __restrict__ dst, const float* __restrict__ src){
  int i = blockIdx.x * 256 + threadIdx.x;
  dst[i] = src[i];
}

// ---------------- host ----------------

static inline void launch_gemm(const float* A, int lda, const float* Bm, int ldb,
                               const float* bias, const float* resid,
                               float* C, int ldc, int M, int N, int K, int act,
                               hipStream_t stream){
  dim3 grid(N / 64, M / 64);
  k_gemm<<<grid, 256, 0, stream>>>(A, lda, Bm, ldb, bias, resid, C, ldc, K, act);
}

extern "C" void kernel_launch(void* const* d_in, const int* in_sizes, int n_in,
                              void* d_out, int out_size, void* d_ws, size_t ws_size,
                              hipStream_t stream){
  (void)in_sizes; (void)n_in; (void)out_size; (void)ws_size;
  const int*   ids  = (const int*)  d_in[0];
  const float* we   = (const float*)d_in[1];
  const float* pe   = (const float*)d_in[2];
  const float* elw  = (const float*)d_in[3];
  const float* elb  = (const float*)d_in[4];
  const float* ist  = (const float*)d_in[5];
  const float* ln1w = (const float*)d_in[6];
  const float* ln1b = (const float*)d_in[7];
  const float* Wqkv = (const float*)d_in[8];
  const float* bqkv = (const float*)d_in[9];
  const float* Wo   = (const float*)d_in[10];
  const float* bo   = (const float*)d_in[11];
  const float* ln2w = (const float*)d_in[12];
  const float* ln2b = (const float*)d_in[13];
  const float* saWq = (const float*)d_in[14];
  const float* sabq = (const float*)d_in[15];
  const float* saWk = (const float*)d_in[16];
  const float* sabk = (const float*)d_in[17];
  const float* saWv = (const float*)d_in[18];
  const float* sabv = (const float*)d_in[19];
  const float* saWo = (const float*)d_in[20];
  const float* sabo = (const float*)d_in[21];
  const float* gWz  = (const float*)d_in[22];
  const float* gbz  = (const float*)d_in[23];
  const float* gWr  = (const float*)d_in[24];
  const float* gbr  = (const float*)d_in[25];
  const float* gWh  = (const float*)d_in[26];
  const float* gbh  = (const float*)d_in[27];
  const float* ln3w = (const float*)d_in[28];
  const float* ln3b = (const float*)d_in[29];
  const float* W1   = (const float*)d_in[30];
  const float* b1   = (const float*)d_in[31];
  const float* W2   = (const float*)d_in[32];
  const float* b2   = (const float*)d_in[33];
  const float* finw = (const float*)d_in[34];
  const float* finb = (const float*)d_in[35];

  float* Wp = (float*)d_ws;
  size_t o = 0;
  float* h    = Wp + o; o += (size_t)BSc * Dc;
  float* xb   = Wp + o; o += (size_t)BSc * Dc;
  float* qkv  = Wp + o; o += (size_t)BSc * D3c;
  float* abuf = Wp + o; o += (size_t)BSc * Dc;
  float* q2   = Wp + o; o += (size_t)BSc * Dc;
  float* sob  = Wp + o; o += (size_t)BSc * Dc;
  float* ffn1 = Wp + o; o += (size_t)BSc * Fic;
  float* ksb  = Wp + o; o += (size_t)BNc * Dc;
  float* vsb  = Wp + o; o += (size_t)BNc * Dc;
  float* st   = Wp + o; o += (size_t)BNc * DSc;
  float* zb   = Wp + o; o += (size_t)BNc * DSc;
  float* rb   = Wp + o; o += (size_t)BNc * DSc;
  float* hcb  = Wp + o; o += (size_t)BNc * DSc;
  float* catb = Wp + o; o += (size_t)BNc * D2c;
  float* cat2 = Wp + o; o += (size_t)BNc * D2c;
  float* sib  = Wp + o; o += (size_t)Bc * Dc;

  k_embed_ln<<<BSc, 256, 0, stream>>>(ids, we, pe, elw, elb, h);
  k_bcast_st<<<(BNc * DSc) / 256, 256, 0, stream>>>(st, ist);

  for (int l = 0; l < Lc; ++l){
    // causal MHA
    k_ln<<<BSc, 256, 0, stream>>>(h, ln1w + l * Dc, ln1b + l * Dc, xb);
    launch_gemm(xb, Dc, Wqkv + (size_t)l * Dc * D3c, D3c, bqkv + l * D3c, nullptr,
                qkv, D3c, BSc, D3c, Dc, 0, stream);
    k_attn<<<(Bc * Hc * (Sc / 16)) / 8, 256, 0, stream>>>(qkv, abuf);
    launch_gemm(abuf, Dc, Wo + (size_t)l * Dc * Dc, Dc, bo + l * Dc, h,
                h, Dc, BSc, Dc, Dc, 0, stream);
    // sparse top-k state attention
    k_ln<<<BSc, 256, 0, stream>>>(h, ln2w + l * Dc, ln2b + l * Dc, xb);
    launch_gemm(xb, Dc, saWq + (size_t)l * Dc * Dc, Dc, sabq + l * Dc, nullptr,
                q2, Dc, BSc, Dc, Dc, 0, stream);
    launch_gemm(st, DSc, saWk + (size_t)l * DSc * Dc, Dc, sabk + l * Dc, nullptr,
                ksb, Dc, BNc, Dc, DSc, 0, stream);
    launch_gemm(st, DSc, saWv + (size_t)l * DSc * Dc, Dc, sabv + l * Dc, nullptr,
                vsb, Dc, BNc, Dc, DSc, 0, stream);
    k_state_attn<<<(Bc * Hc * Sc) / 8, 256, 0, stream>>>(q2, ksb, vsb, sob);
    launch_gemm(sob, Dc, saWo + (size_t)l * Dc * Dc, Dc, sabo + l * Dc, h,
                h, Dc, BSc, Dc, Dc, 0, stream);
    // GRU state propagation
    k_mean_seq<<<(Bc * Dc) / 256, 256, 0, stream>>>(h, sib);
    k_cat<<<(BNc * D2c) / 256, 256, 0, stream>>>(st, sib, catb);
    launch_gemm(catb, D2c, gWz + (size_t)l * D2c * DSc, DSc, gbz + l * DSc, nullptr,
                zb, DSc, BNc, DSc, D2c, 2, stream);
    launch_gemm(catb, D2c, gWr + (size_t)l * D2c * DSc, DSc, gbr + l * DSc, nullptr,
                rb, DSc, BNc, DSc, D2c, 2, stream);
    k_cat2<<<(BNc * D2c) / 256, 256, 0, stream>>>(st, sib, rb, cat2);
    launch_gemm(cat2, D2c, gWh + (size_t)l * D2c * DSc, DSc, gbh + l * DSc, nullptr,
                hcb, DSc, BNc, DSc, D2c, 3, stream);
    k_gru<<<(BNc * DSc) / 256, 256, 0, stream>>>(st, zb, hcb);
    // FFN
    k_ln<<<BSc, 256, 0, stream>>>(h, ln3w + l * Dc, ln3b + l * Dc, xb);
    launch_gemm(xb, Dc, W1 + (size_t)l * Dc * Fic, Fic, b1 + l * Fic, nullptr,
                ffn1, Fic, BSc, Fic, Dc, 1, stream);
    launch_gemm(ffn1, Fic, W2 + (size_t)l * Fic * Dc, Dc, b2 + l * Dc, h,
                h, Dc, BSc, Dc, Fic, 0, stream);
  }

  float* out = (float*)d_out;
  k_ln<<<BSc, 256, 0, stream>>>(h, finw, finb, out);
  k_copy<<<(BNc * DSc) / 256, 256, 0, stream>>>(out + (size_t)BSc * Dc, st);
}
